// PoseCDE_90829968375899
// MI455X (gfx1250) — matrix operand order repaired and checked by reading, not compiled, not run on hardware
//
#include <hip/hip_runtime.h>
#include <math.h>

typedef __attribute__((ext_vector_type(16))) __bf16 v16bf;
typedef __attribute__((ext_vector_type(8)))  float  v8f;
typedef __attribute__((ext_vector_type(4)))  unsigned int uint32x4;
typedef __attribute__((ext_vector_type(8)))  int int32x8;
typedef __attribute__((ext_vector_type(4)))  int int32x4;

#define B_SZ   4096
#define H_DIM  128
#define F_DIM  129
#define KRED   768
#define OUTN   (H_DIM * F_DIM)   /* 16512 */
#define NSTEPS 36
#define HSTEP  0.025f
#define NEVAL  10
#define MROWS  ((B_SZ) * (NEVAL)) /* 40960 */

#define WSTR   132   /* LDS weight row stride (halves): 66 dwords -> bank-conflict-free */
#define ASTR   136   /* LDS activation row stride (halves): 68 dwords, 16B-aligned rows */

/* LDS layout (bytes) */
#define SM_W     0
#define SM_WSZ   (4 * 128 * WSTR * 2)            /* 135168 */
#define SM_ZT    (SM_W + SM_WSZ)                 /* ztmp   16*ASTR*2 = 4352 */
#define SM_HB    (SM_ZT + 16 * ASTR * 2)         /* hdnB   4352 */
#define SM_HA    (SM_HB + 16 * ASTR * 2)         /* hdnA   4352 */
#define SM_D1    (SM_HA + 16 * ASTR * 2)         /* d1s    8192 */
#define SM_DT    (SM_D1 + 16 * H_DIM * 4)        /* dts    64   */
#define SM_TOT   (SM_DT + 64)                    /* 156480 */

__device__ __forceinline__ float fast_tanh(float x) {
#if __has_builtin(__builtin_amdgcn_tanhf)
  return __builtin_amdgcn_tanhf(x);   /* gfx1250 v_tanh_f32 (TRANS unit) */
#else
  return tanhf(x);
#endif
}

// ---------------- WMMA fragment helpers (CDNA5 wave32 layouts) ----------------
// A (16x32 bf16): lane<16 -> row=lane, K {0..7,16..23}; lane>=16 -> row=lane-16, K {8..15,24..31}
__device__ __forceinline__ v16bf load_a16(const __bf16* A, int lda, int kc, int lane) {
  const int row = lane & 15;
  const int khi = (lane >> 4) & 1;
  const __bf16* p = A + row * lda + kc * 32 + khi * 8;
  v16bf a;
#pragma unroll
  for (int i = 0; i < 8; ++i) { a[i] = p[i]; a[i + 8] = p[16 + i]; }
  return a;
}
// B (32x16 bf16): lane<16 -> col=lane, K {0..7,16..23}; lane>=16 -> col=lane-16, K {8..15,24..31}
__device__ __forceinline__ v16bf load_b16(const __bf16* W, int ldw, int kc, int n0, int lane) {
  const __bf16* p = W + (size_t)(kc * 32 + ((lane >> 4) & 1) * 8) * ldw + n0 + (lane & 15);
  v16bf b;
#pragma unroll
  for (int i = 0; i < 8; ++i) { b[i] = p[(size_t)i * ldw]; b[i + 8] = p[(size_t)(16 + i) * ldw]; }
  return b;
}
__device__ __forceinline__ v8f wmma_bf16(v16bf a, v16bf b, v8f c) {
  return __builtin_amdgcn_wmma_f32_16x16x32_bf16(false, a, false, b, (short)0, c, false, false);
}
// C/D (16x16 f32): elem i -> row = i + 8*(lane>=16), col = (lane&15) + n0

// 16x128 tile: dst = tanh(src @ W + bias)
__device__ __forceinline__ void tile_gemm_tanh(const __bf16* src, int lda,
                                               const __bf16* W, int ldw,
                                               const float* bias, __bf16* dst, int ldd,
                                               int lane, int wave) {
  const int rbase = ((lane >> 4) & 1) << 3;
#pragma unroll
  for (int t = 0; t < 2; ++t) {
    const int n0 = wave * 32 + t * 16;
    v8f acc = {};
#pragma unroll
    for (int kc = 0; kc < 4; ++kc) {
      v16bf a = load_a16(src, lda, kc, lane);
      v16bf b = load_b16(W, ldw, kc, n0, lane);
      acc = wmma_bf16(a, b, acc);
    }
    const int col = n0 + (lane & 15);
    const float bv = bias[col];
#pragma unroll
    for (int i = 0; i < 8; ++i)
      dst[(rbase + i) * ldd + col] = (__bf16)fast_tanh(acc[i] + bv);
  }
}

// k = tanh(hdn @ Wc0 + bc0) * dt[row]
__device__ __forceinline__ void col0_stage(const __bf16* hdn, int lda,
                                           const __bf16* Wc0, int ldw,
                                           const float* bc0, const float* dts,
                                           float kv[2][8], int lane, int wave) {
  const int rbase = ((lane >> 4) & 1) << 3;
#pragma unroll
  for (int t = 0; t < 2; ++t) {
    const int n0 = wave * 32 + t * 16;
    v8f acc = {};
#pragma unroll
    for (int kc = 0; kc < 4; ++kc) {
      v16bf a = load_a16(hdn, lda, kc, lane);
      v16bf b = load_b16(Wc0, ldw, kc, n0, lane);
      acc = wmma_bf16(a, b, acc);
    }
    const int col = n0 + (lane & 15);
    const float bv = bc0[col];
#pragma unroll
    for (int i = 0; i < 8; ++i)
      kv[t][i] = fast_tanh(acc[i] + bv) * dts[rbase + i];
  }
}

__device__ __forceinline__ void mlp_hidden(const __bf16* zin, __bf16* hA, __bf16* hB,
                                           const __bf16* Win, const __bf16* Wh0,
                                           const __bf16* Wh1, const float* b_in,
                                           const float* bh0, const float* bh1,
                                           int lane, int wave) {
  tile_gemm_tanh(zin, ASTR, Win, WSTR, b_in, hA, ASTR, lane, wave); __syncthreads();
  tile_gemm_tanh(hA,  ASTR, Wh0, WSTR, bh0,  hB, ASTR, lane, wave); __syncthreads();
  tile_gemm_tanh(hB,  ASTR, Wh1, WSTR, bh1,  hA, ASTR, lane, wave); __syncthreads();
}

// Final k4: full 16x16512 W_out pass, contracted over f>=1 with d1 features.
__device__ __forceinline__ void full_stage(const __bf16* hdn, const __bf16* Wout,
                                           const float* b_out, const float* d1s,
                                           float* kbuf, float kv[2][8],
                                           int tid, int lane, int wave) {
  for (int i = tid; i < 16 * H_DIM; i += 128) kbuf[i] = 0.f;
  __syncthreads();
  const int rbase = ((lane >> 4) & 1) << 3;
  for (int nt = wave; nt < OUTN / 16; nt += 4) {     // 1032 N-tiles / 4 waves
    const int n0 = nt * 16;
    v8f acc = {};
#pragma unroll
    for (int kc = 0; kc < 4; ++kc) {
      v16bf a = load_a16(hdn, ASTR, kc, lane);
      v16bf b = load_b16(Wout, OUTN, kc, n0, lane);
      acc = wmma_bf16(a, b, acc);
    }
    const int col = n0 + (lane & 15);
    const int hh = col / F_DIM;
    const int ff = col - hh * F_DIM;
    if (ff != 0) {                                   // f=0 column multiplies zero (time ch of d1)
      const float bv = b_out[col];
#pragma unroll
      for (int i = 0; i < 8; ++i) {
        const int r = rbase + i;
        atomicAdd(&kbuf[r * H_DIM + hh], fast_tanh(acc[i] + bv) * d1s[r * H_DIM + (ff - 1)]);
      }
    }
  }
  __syncthreads();
#pragma unroll
  for (int t = 0; t < 2; ++t) {
    const int col = wave * 32 + t * 16 + (lane & 15);
#pragma unroll
    for (int i = 0; i < 8; ++i)
      kv[t][i] = kbuf[(rbase + i) * H_DIM + col];
  }
  __syncthreads();   // close kbuf alias window before ztmp is rewritten
}

// ---------------- kernels ----------------
__global__ __launch_bounds__(256) void cvt_weights_kernel(
    const float* Wred, const float* Win, const float* Wh, const float* Wout,
    const float* Wr1, const float* bout,
    __bf16* Wred_b, __bf16* Wfused_b, __bf16* Wout_b, __bf16* Wr1_b, float* bc0) {
  const int stride = gridDim.x * blockDim.x;
  for (int i = blockIdx.x * blockDim.x + threadIdx.x; i < KRED * H_DIM; i += stride)
    Wred_b[i] = (__bf16)Wred[i];
  for (int i = blockIdx.x * blockDim.x + threadIdx.x; i < H_DIM * H_DIM; i += stride) {
    Wfused_b[i]                   = (__bf16)Win[i];                 // W_in
    Wfused_b[1 * H_DIM * H_DIM + i] = (__bf16)Wh[i];                // W_h0
    Wfused_b[2 * H_DIM * H_DIM + i] = (__bf16)Wh[H_DIM * H_DIM + i];// W_h1
    Wr1_b[i] = (__bf16)Wr1[i];
    const int k = i >> 7, h = i & 127;
    Wfused_b[3 * H_DIM * H_DIM + i] = (__bf16)Wout[(size_t)k * OUTN + h * F_DIM]; // Wc0
  }
  for (size_t i = blockIdx.x * blockDim.x + threadIdx.x; i < (size_t)H_DIM * OUTN; i += stride)
    Wout_b[i] = (__bf16)Wout[i];
  for (int i = blockIdx.x * blockDim.x + threadIdx.x; i < H_DIM; i += stride)
    bc0[i] = bout[i * F_DIM];
}

__global__ __launch_bounds__(256) void prep_kernel(
    const float* fv, const float* fi, const float* ts, __bf16* xd, float* dt0) {
  const int stride = gridDim.x * blockDim.x;
  for (int i = blockIdx.x * blockDim.x + threadIdx.x; i < B_SZ * KRED; i += stride) {
    const int b = i / KRED, k = i - b * KRED;
    float a1, a0;
    if (k < 512) { a1 = fv[(b * 10 + 1) * 512 + k];       a0 = fv[(b * 10 + 0) * 512 + k]; }
    else         { a1 = fi[(b * 10 + 1) * 256 + (k-512)]; a0 = fi[(b * 10 + 0) * 256 + (k-512)]; }
    xd[i] = (__bf16)(a1 - a0);
  }
  for (int i = blockIdx.x * blockDim.x + threadIdx.x; i < B_SZ; i += stride)
    dt0[i] = ts[i * 11 + 2] - ts[i * 11 + 1];
}

// d1feat0 = (X[:,1]-X[:,0]) @ W_red   (bias cancels)
__global__ __launch_bounds__(128) void d1feat_gemm_kernel(
    const __bf16* __restrict__ Xd, const __bf16* __restrict__ Wred,
    float* __restrict__ out) {
  const int lane = threadIdx.x & 31, wave = threadIdx.x >> 5;
  const int m0 = blockIdx.x * 16;
  const int rbase = ((lane >> 4) & 1) << 3;
#pragma unroll
  for (int t = 0; t < 2; ++t) {
    const int n0 = wave * 32 + t * 16;
    v8f acc = {};
    for (int kc = 0; kc < KRED / 32; ++kc) {
      v16bf a = load_a16(Xd + (size_t)m0 * KRED, KRED, kc, lane);
      v16bf b = load_b16(Wred, H_DIM, kc, n0, lane);
      acc = wmma_bf16(a, b, acc);
    }
    const int col = n0 + (lane & 15);
#pragma unroll
    for (int i = 0; i < 8; ++i)
      out[(m0 + rbase + i) * H_DIM + col] = acc[i];
  }
}

__global__ __launch_bounds__(128) void cde_layer_kernel(
    const float* __restrict__ d1feat, const float* __restrict__ dt0,
    const __bf16* __restrict__ Wfused,   /* Win|Wh0|Wh1|Wc0 : 512 x 128 row-major bf16 */
    const __bf16* __restrict__ Wout,
    const float* __restrict__ b_in, const float* __restrict__ bh0,
    const float* __restrict__ bh1, const float* __restrict__ bc0,
    const float* __restrict__ b_out, float* __restrict__ h_out) {
  __shared__ __attribute__((aligned(16))) unsigned char smem[SM_TOT];
  __bf16* wlds = (__bf16*)(smem + SM_W);
  __bf16* ztmp = (__bf16*)(smem + SM_ZT);
  __bf16* hdnB = (__bf16*)(smem + SM_HB);
  __bf16* hdnA = (__bf16*)(smem + SM_HA);
  float*  d1s  = (float*)(smem + SM_D1);
  float*  dts  = (float*)(smem + SM_DT);
  float*  kbuf = (float*)(smem + SM_ZT);   // aliases ztmp+hdnB (dead during final stage)
  const __bf16* Win = wlds;
  const __bf16* Wh0 = wlds + 128 * WSTR;
  const __bf16* Wh1 = wlds + 256 * WSTR;
  const __bf16* Wc0 = wlds + 384 * WSTR;

  const int tid = threadIdx.x;
  const int lane = tid & 31;
  const int wave = tid >> 5;
  const int b0 = blockIdx.x * 16;
  const int rbase = ((lane >> 4) & 1) << 3;

  __builtin_prefetch(Wout, 0, 1);   // warm L2 with the streamed final-stage weights

  // ---- stage the 4 hidden-layer weight matrices (512x128 bf16) into padded LDS ----
#if __has_builtin(__builtin_amdgcn_tensor_load_to_lds) && (__clang_major__ < 23)
  if (wave == 0) {
    // Tensor DMA: 2-D tile 512 rows x 128 halves, data_size=2B,
    // pad_enable, pad_interval=5 (64 dwords = one row), pad_amount=1 (2 dwords)
    // -> LDS row stride 66 dwords = 132 halves = WSTR.
    unsigned long long ga = (unsigned long long)(size_t)Wfused;
    unsigned int ldsa = (unsigned int)(size_t)wlds;   // low 32 bits = LDS offset
    uint32x4 g0;
    g0[0] = 1u;                                            // count=1, user descriptor
    g0[1] = ldsa;                                          // lds_addr
    g0[2] = (unsigned int)ga;                              // global_addr[31:0]
    g0[3] = ((unsigned int)(ga >> 32) & 0x01ffffffu) | (2u << 30);  // addr[56:32] | type=2
    int32x8 g1;
    g1[0] = (1 << 16) | (1 << 20) | (5 << 22) | (1 << 25); // data_size=2B, pad en/intvl/amt
    g1[1] = (128 & 0xffff) << 16;                          // tensor_dim0 = 128 (lo16)
    g1[2] = (512 << 16);                                   // dim0 hi=0 | tensor_dim1=512 lo16
    g1[3] = (128 << 16);                                   // dim1 hi=0 | tile_dim0=128
    g1[4] = 512;                                           // tile_dim1=512 | tile_dim2=0
    g1[5] = 128;                                           // tensor_dim0_stride = 128
    g1[6] = 0;
    g1[7] = 0;
    int32x4 g2 = {0, 0, 0, 0};
    int32x4 g3 = {0, 0, 0, 0};
    __builtin_amdgcn_tensor_load_to_lds(g0, g1, g2, g3, 0);
    __builtin_amdgcn_s_wait_tensorcnt(0);
  }
#else
  for (int i = tid; i < 512 * H_DIM; i += 128) {
    const int r = i >> 7, c = i & 127;
    wlds[r * WSTR + c] = Wfused[i];
  }
#endif

  for (int i = tid; i < 16 * H_DIM; i += 128) {
    const int r = i >> 7, c = i & 127;
    d1s[i] = d1feat[(b0 + r) * H_DIM + c];
    h_out[((b0 + r) * NEVAL + 0) * H_DIM + c] = 0.f;  // h_i[:,0] = z0 = 0
    ztmp[r * ASTR + c] = (__bf16)0.f;
  }
  if (tid < 16) dts[tid] = dt0[b0 + tid];
  __syncthreads();

  float zcur[2][8];
#pragma unroll
  for (int t = 0; t < 2; ++t)
#pragma unroll
    for (int i = 0; i < 8; ++i) zcur[t][i] = 0.f;

  for (int step = 0; step < NSTEPS; ++step) {
    float zacc[2][8];
    float kv[2][8];

    // k1
    mlp_hidden(ztmp, hdnA, hdnB, Win, Wh0, Wh1, b_in, bh0, bh1, lane, wave);
    col0_stage(hdnA, ASTR, Wc0, WSTR, bc0, dts, kv, lane, wave);
#pragma unroll
    for (int t = 0; t < 2; ++t) {
      const int col = wave * 32 + t * 16 + (lane & 15);
#pragma unroll
      for (int i = 0; i < 8; ++i) {
        zacc[t][i] = kv[t][i];
        ztmp[(rbase + i) * ASTR + col] = (__bf16)(zcur[t][i] + 0.5f * HSTEP * kv[t][i]);
      }
    }
    __syncthreads();

    // k2
    mlp_hidden(ztmp, hdnA, hdnB, Win, Wh0, Wh1, b_in, bh0, bh1, lane, wave);
    col0_stage(hdnA, ASTR, Wc0, WSTR, bc0, dts, kv, lane, wave);
#pragma unroll
    for (int t = 0; t < 2; ++t) {
      const int col = wave * 32 + t * 16 + (lane & 15);
#pragma unroll
      for (int i = 0; i < 8; ++i) {
        zacc[t][i] += 2.f * kv[t][i];
        ztmp[(rbase + i) * ASTR + col] = (__bf16)(zcur[t][i] + 0.5f * HSTEP * kv[t][i]);
      }
    }
    __syncthreads();

    // k3
    mlp_hidden(ztmp, hdnA, hdnB, Win, Wh0, Wh1, b_in, bh0, bh1, lane, wave);
    col0_stage(hdnA, ASTR, Wc0, WSTR, bc0, dts, kv, lane, wave);
#pragma unroll
    for (int t = 0; t < 2; ++t) {
      const int col = wave * 32 + t * 16 + (lane & 15);
#pragma unroll
      for (int i = 0; i < 8; ++i) {
        zacc[t][i] += 2.f * kv[t][i];
        ztmp[(rbase + i) * ASTR + col] = (__bf16)(zcur[t][i] + HSTEP * kv[t][i]);
      }
    }
    __syncthreads();

    // k4 (the one at t==1.0 uses the feature-segment deriv -> full W_out pass)
    mlp_hidden(ztmp, hdnA, hdnB, Win, Wh0, Wh1, b_in, bh0, bh1, lane, wave);
    if (step == NSTEPS - 1)
      full_stage(hdnA, Wout, b_out, d1s, kbuf, kv, tid, lane, wave);
    else
      col0_stage(hdnA, ASTR, Wc0, WSTR, bc0, dts, kv, lane, wave);

#pragma unroll
    for (int t = 0; t < 2; ++t) {
      const int col = wave * 32 + t * 16 + (lane & 15);
#pragma unroll
      for (int i = 0; i < 8; ++i) {
        zcur[t][i] += (HSTEP / 6.f) * (zacc[t][i] + kv[t][i]);
        ztmp[(rbase + i) * ASTR + col] = (__bf16)zcur[t][i];
      }
    }
    if (((step + 1) & 3) == 0) {
      const int e = (step + 1) >> 2;
#pragma unroll
      for (int t = 0; t < 2; ++t) {
        const int col = wave * 32 + t * 16 + (lane & 15);
#pragma unroll
        for (int i = 0; i < 8; ++i)
          h_out[((b0 + rbase + i) * NEVAL + e) * H_DIM + col] = zcur[t][i];
      }
    }
    __syncthreads();
  }
}

__global__ __launch_bounds__(256) void d1next_kernel(const float* hA, float* d1f) {
  const int stride = gridDim.x * blockDim.x;
  for (int i = blockIdx.x * blockDim.x + threadIdx.x; i < B_SZ * H_DIM; i += stride) {
    const int b = i >> 7, c = i & 127;
    d1f[i] = hA[(b * NEVAL + 1) * H_DIM + c] - hA[(b * NEVAL + 0) * H_DIM + c];
  }
}

__global__ __launch_bounds__(256) void hdiff_kernel(const float* hB, __bf16* Hd) {
  const int stride = gridDim.x * blockDim.x;
  for (int i = blockIdx.x * blockDim.x + threadIdx.x; i < B_SZ * NEVAL * H_DIM; i += stride) {
    const int b = i / (NEVAL * H_DIM);
    const int rem = i - b * (NEVAL * H_DIM);
    const int e = rem >> 7, c = rem & 127;
    float v = hB[(b * NEVAL + e) * H_DIM + c];
    if (e > 0) v -= hB[(b * NEVAL + e - 1) * H_DIM + c];
    Hd[i] = (__bf16)v;
  }
}

__global__ __launch_bounds__(128) void head_r1_kernel(
    const __bf16* __restrict__ Hd, const __bf16* __restrict__ Wr1,
    const float* __restrict__ br1, float* __restrict__ hid) {
  const int lane = threadIdx.x & 31, wave = threadIdx.x >> 5;
  const int m0 = blockIdx.x * 16;
  const int rbase = ((lane >> 4) & 1) << 3;
#pragma unroll
  for (int t = 0; t < 2; ++t) {
    const int n0 = wave * 32 + t * 16;
    v8f acc = {};
#pragma unroll
    for (int kc = 0; kc < 4; ++kc) {
      v16bf a = load_a16(Hd + (size_t)m0 * H_DIM, H_DIM, kc, lane);
      v16bf b = load_b16(Wr1, H_DIM, kc, n0, lane);
      acc = wmma_bf16(a, b, acc);
    }
    const int col = n0 + (lane & 15);
    const float bv = br1[col];
#pragma unroll
    for (int i = 0; i < 8; ++i) {
      float v = acc[i] + bv;
      hid[(m0 + rbase + i) * H_DIM + col] = (v > 0.f) ? v : 0.1f * v;
    }
  }
}

__global__ __launch_bounds__(256) void head_r2_kernel(
    const float* hid, const float* Wr2, const float* br2, float* poses) {
  const int stride = gridDim.x * blockDim.x;
  for (int i = blockIdx.x * blockDim.x + threadIdx.x; i < MROWS * 6; i += stride) {
    const int row = i / 6, j = i - row * 6;
    float s = br2[j];
    for (int k = 0; k < H_DIM; ++k) s += hid[row * H_DIM + k] * Wr2[k * 6 + j];
    poses[i] = s;
  }
}

__global__ __launch_bounds__(256) void hT_kernel(const float* hA, const float* hB, float* out) {
  const int stride = gridDim.x * blockDim.x;
  for (int i = blockIdx.x * blockDim.x + threadIdx.x; i < 2 * B_SZ * H_DIM; i += stride) {
    const int l = i / (B_SZ * H_DIM);
    const int r = i - l * (B_SZ * H_DIM);
    const int b = r >> 7, c = r & 127;
    const float* src = l ? hB : hA;
    out[i] = src[(b * NEVAL + (NEVAL - 1)) * H_DIM + c];
  }
}

// ---------------- host launch ----------------
extern "C" void kernel_launch(void* const* d_in, const int* in_sizes, int n_in,
                              void* d_out, int out_size, void* d_ws, size_t ws_size,
                              hipStream_t stream) {
  const float* fv    = (const float*)d_in[0];
  const float* fi    = (const float*)d_in[1];
  const float* ts    = (const float*)d_in[2];
  const float* W_red = (const float*)d_in[3];
  const float* W_in  = (const float*)d_in[5];
  const float* b_in  = (const float*)d_in[6];
  const float* W_h   = (const float*)d_in[7];
  const float* b_h   = (const float*)d_in[8];
  const float* W_out = (const float*)d_in[9];
  const float* b_out = (const float*)d_in[10];
  const float* W_r1  = (const float*)d_in[11];
  const float* b_r1  = (const float*)d_in[12];
  const float* W_r2  = (const float*)d_in[13];
  const float* b_r2  = (const float*)d_in[14];
  (void)in_sizes; (void)n_in; (void)out_size; (void)ws_size;

  char* base = (char*)d_ws;
  size_t off = 0;
  auto wsalloc = [&](size_t bytes) -> void* {
    off = (off + 255) & ~(size_t)255;
    void* p = base + off;
    off += bytes;
    return p;
  };
  __bf16* Wred_b  = (__bf16*)wsalloc((size_t)KRED * H_DIM * 2);
  __bf16* Wfused_b= (__bf16*)wsalloc((size_t)4 * H_DIM * H_DIM * 2); /* Win|Wh0|Wh1|Wc0 */
  __bf16* Wr1_b   = (__bf16*)wsalloc((size_t)H_DIM * H_DIM * 2);
  __bf16* Wout_b  = (__bf16*)wsalloc((size_t)H_DIM * OUTN * 2);
  float*  bc0     = (float*) wsalloc((size_t)H_DIM * 4);
  __bf16* xd      = (__bf16*)wsalloc((size_t)B_SZ * KRED * 2);
  float*  dt0     = (float*) wsalloc((size_t)B_SZ * 4);
  float*  d1f0    = (float*) wsalloc((size_t)B_SZ * H_DIM * 4);
  float*  d1f1    = (float*) wsalloc((size_t)B_SZ * H_DIM * 4);
  float*  hA      = (float*) wsalloc((size_t)B_SZ * NEVAL * H_DIM * 4);
  float*  hB      = (float*) wsalloc((size_t)B_SZ * NEVAL * H_DIM * 4);
  __bf16* Hd      = (__bf16*)wsalloc((size_t)MROWS * H_DIM * 2);
  float*  hid     = (float*) wsalloc((size_t)MROWS * H_DIM * 4);

  cvt_weights_kernel<<<2048, 256, 0, stream>>>(W_red, W_in, W_h, W_out, W_r1, b_out,
                                               Wred_b, Wfused_b, Wout_b, Wr1_b, bc0);
  prep_kernel<<<2048, 256, 0, stream>>>(fv, fi, ts, xd, dt0);
  d1feat_gemm_kernel<<<B_SZ / 16, 128, 0, stream>>>(xd, Wred_b, d1f0);

  cde_layer_kernel<<<B_SZ / 16, 128, 0, stream>>>(d1f0, dt0, Wfused_b, Wout_b,
                                                  b_in, b_h, b_h + H_DIM, bc0, b_out, hA);
  d1next_kernel<<<1024, 256, 0, stream>>>(hA, d1f1);
  cde_layer_kernel<<<B_SZ / 16, 128, 0, stream>>>(d1f1, dt0, Wfused_b, Wout_b,
                                                  b_in, b_h, b_h + H_DIM, bc0, b_out, hB);

  hdiff_kernel<<<2048, 256, 0, stream>>>(hB, Hd);
  head_r1_kernel<<<MROWS / 16, 128, 0, stream>>>(Hd, Wr1_b, b_r1, hid);

  float* poses = (float*)d_out;
  head_r2_kernel<<<2048, 256, 0, stream>>>(hid, W_r2, b_r2, poses);
  hT_kernel<<<1024, 256, 0, stream>>>(hA, hB, poses + (size_t)MROWS * 6);
}
